// WaterNet1116_16690242913229
// MI455X (gfx1250) — matrix operand order repaired and checked by reading, not compiled, not run on hardware
//
#include <hip/hip_runtime.h>
#include <hip/hip_bf16.h>
#include <math.h>

#define NT 730
#define NS 2000
#define NH 16
#define NG 32
#define NR 15

typedef __attribute__((ext_vector_type(16))) _Float16 v16h;
typedef __attribute__((ext_vector_type(8)))  float    v8f;

// ---------------------------------------------------------------------------
// Fast transcendental helpers
// ---------------------------------------------------------------------------
__device__ __forceinline__ float fast_tanh(float x) {
#if __has_builtin(__builtin_amdgcn_tanhf)
  return __builtin_amdgcn_tanhf(x);
#elif __has_builtin(__builtin_amdgcn_tanh_f32)
  return __builtin_amdgcn_tanh_f32(x);
#else
  // CDNA5 hardware TANH trans op; v_nop covers the TRANS result-use hazard
  float r;
  asm volatile("v_tanh_f32 %0, %1\n\tv_nop" : "=v"(r) : "v"(x));
  return r;
#endif
}

__device__ __forceinline__ float sigm(float x) { return 1.0f / (1.0f + __expf(-x)); }
__device__ __forceinline__ float hsig(float z) {
  return fminf(1.0f, fmaxf(0.0f, z * (1.0f / 6.0f) + 0.5f));
}

// ---------------------------------------------------------------------------
// WMMA fragment helpers (CDNA5 wave32 layouts, CDNA5 ISA §7.12.2)
//
// B fragments are PRE-PACKED in per-lane register order by prep_weights_kernel:
//   Bp[fi*512 + lane*16 + idx]  ==  halves[idx] of fragment fi for this lane
// so a fragment load is one contiguous 32-byte vector load per lane.
// ---------------------------------------------------------------------------
__device__ __forceinline__ v16h load_frag_packed(const _Float16* __restrict__ Bp,
                                                 int fi) {
  const v16h* __restrict__ vp = (const v16h*)Bp;
  return vp[fi * 32 + (threadIdx.x & 31)];
}

// A fragment (M=16 x K=32) from LDS stage [16][32] f16 row-major.
// lane M=lane&15; half=lane>>4: halves[0..7]=K=half*8+i, halves[8..15]=K=16+half*8+i
// (two contiguous 8-half runs per lane -> ds_load_b128 pairs)
__device__ __forceinline__ v16h load_a_stage(const _Float16* __restrict__ S) {
  const int lane = threadIdx.x & 31;
  const int M    = lane & 15;
  const int hf   = (lane >> 4) * 8;
  v16h a;
#pragma unroll
  for (int i = 0; i < 8; ++i) {
    a[i]     = S[M * 32 + hf + i];
    a[8 + i] = S[M * 32 + 16 + hf + i];
  }
  return a;
}

// ---------------------------------------------------------------------------
// Kernel 1: pack fcT* weights to f16 fragment-ordered buffers (zero K-padding).
//   B1a: 32 frags, fi = n1*2 + kstep   (layer1 fcT1, K=64 pad of 33, N=256)
//   B2a: 16 frags, fi = kk*2 + n2     (layer2 fcT1, K=256, N=32)
//   B1b: 32 frags, fi = n1*2 + kstep  (layer1 fcT2, K=64 pad of 35, N=256)
//   B2b:  8 frags, fi = kk            (layer2 fcT2, K=256, N=16 of 17 used)
// B-frag lane layout: col = lane&15; half=lane>>4: idx 0-7 K=kbase+half*8+idx, idx 8-15 K=kbase+16+half*8+idx-8
// ---------------------------------------------------------------------------
__global__ void prep_weights_kernel(const float* __restrict__ fcT1W1,  // [256,33]
                                    const float* __restrict__ fcT1W2,  // [32,256]
                                    const float* __restrict__ fcT2W1,  // [256,35]
                                    const float* __restrict__ fcT2W2,  // [17,256]
                                    _Float16* __restrict__ B1a,
                                    _Float16* __restrict__ B2a,
                                    _Float16* __restrict__ B1b,
                                    _Float16* __restrict__ B2b) {
  const int i      = blockIdx.x * blockDim.x + threadIdx.x;
  const int stride = gridDim.x * blockDim.x;
  for (int p = i; p < 32 * 512; p += stride) {          // B1a
    const int fi = p >> 9, lane = (p >> 4) & 31, idx = p & 15;
    const int n1 = fi >> 1, kstep = fi & 1;
    const int k = kstep * 32 + ((lane >> 4) << 3) + (idx & 7) + ((idx >> 3) << 4);
    const int n = n1 * 16 + (lane & 15);
    B1a[p] = (_Float16)(k < 33 ? fcT1W1[n * 33 + k] : 0.0f);
  }
  for (int p = i; p < 16 * 512; p += stride) {          // B2a
    const int fi = p >> 9, lane = (p >> 4) & 31, idx = p & 15;
    const int kk = fi >> 1, n2 = fi & 1;
    const int k = kk * 32 + ((lane >> 4) << 3) + (idx & 7) + ((idx >> 3) << 4);
    const int n = n2 * 16 + (lane & 15);
    B2a[p] = (_Float16)fcT1W2[n * 256 + k];
  }
  for (int p = i; p < 32 * 512; p += stride) {          // B1b
    const int fi = p >> 9, lane = (p >> 4) & 31, idx = p & 15;
    const int n1 = fi >> 1, kstep = fi & 1;
    const int k = kstep * 32 + ((lane >> 4) << 3) + (idx & 7) + ((idx >> 3) << 4);
    const int n = n1 * 16 + (lane & 15);
    B1b[p] = (_Float16)(k < 35 ? fcT2W1[n * 35 + k] : 0.0f);
  }
  for (int p = i; p < 8 * 512; p += stride) {           // B2b
    const int fi = p >> 9, lane = (p >> 4) & 31, idx = p & 15;
    const int k = fi * 32 + ((lane >> 4) << 3) + (idx & 7) + ((idx >> 3) << 4);
    const int n = lane & 15;                            // first NH of 17 outputs
    B2b[p] = (_Float16)fcT2W2[n * 256 + k];
  }
}

// ---------------------------------------------------------------------------
// Kernel 2: tiny per-basin MLPs fc / fcR  (xc: [NS,32] -> w:[NS,160], wR:[NS,240])
// block = one basin s, 256 threads
// ---------------------------------------------------------------------------
__global__ __launch_bounds__(256) void params_mlp_kernel(
    const float* __restrict__ xc,
    const float* __restrict__ fcW1,  const float* __restrict__ fcb1,
    const float* __restrict__ fcW2,  const float* __restrict__ fcb2,
    const float* __restrict__ fcRW1, const float* __restrict__ fcRb1,
    const float* __restrict__ fcRW2, const float* __restrict__ fcRb2,
    float* __restrict__ wbuf, float* __restrict__ wRbuf) {
  __shared__ float xcs[NG];
  __shared__ float hid[256];
  const int s   = blockIdx.x;
  const int tid = threadIdx.x;
  if (tid < NG) xcs[tid] = xc[s * NG + tid];
  __syncthreads();
  {
    float acc = fcb1[tid];
    const float* wr = fcW1 + tid * NG;
#pragma unroll
    for (int g = 0; g < NG; ++g) acc += xcs[g] * wr[g];
    hid[tid] = fast_tanh(acc);
  }
  __syncthreads();
  if (tid < NH * 10) {
    float acc = fcb2[tid];
    const float* wr = fcW2 + tid * 256;
    for (int j = 0; j < 256; ++j) acc += hid[j] * wr[j];
    wbuf[s * (NH * 10) + tid] = acc;
  }
  __syncthreads();
  {
    float acc = fcRb1[tid];
    const float* wr = fcRW1 + tid * NG;
#pragma unroll
    for (int g = 0; g < NG; ++g) acc += xcs[g] * wr[g];
    hid[tid] = fast_tanh(acc);
  }
  __syncthreads();
  if (tid < NH * NR) {
    float acc = fcRb2[tid];
    const float* wr = fcRW2 + tid * 256;
    for (int j = 0; j < 256; ++j) acc += hid[j] * wr[j];
    wRbuf[s * (NH * NR) + tid] = acc;
  }
}

// ---------------------------------------------------------------------------
// Kernel 3: fused 2-layer MLPs fcT1/fcT2 over all NT*NS rows, f16 WMMA.
// One wave per 16-row tile; hidden staged 32-at-a-time through LDS;
// pre-packed B fragments fetched as single 32B vector loads (L2-hot);
// hardware v_tanh on hidden activations (co-executes with the matrix pipe).
// ---------------------------------------------------------------------------
__global__ __launch_bounds__(256) void fused_mlp_kernel(
    const float* __restrict__ x, const float* __restrict__ xc,
    const _Float16* __restrict__ B1a, const float* __restrict__ b1a,
    const _Float16* __restrict__ B2a, const float* __restrict__ b2a,
    const _Float16* __restrict__ B1b, const float* __restrict__ b1b,
    const _Float16* __restrict__ B2b, const float* __restrict__ b2b,
    float* __restrict__ v1ws, float* __restrict__ v2ws) {
  __shared__ _Float16 stage[8][16 * 32];
  const int wave = threadIdx.x >> 5;
  const int lane = threadIdx.x & 31;
  const int tile = blockIdx.x * 8 + wave;
  const int ntiles = (NT * NS) / 16;   // 91250, exact
  if (tile >= ntiles) return;          // whole-wave uniform exit: WMMA EXEC stays all-1s

  const int row0  = tile * 16;
  const int t     = row0 / NS;         // NS % 16 == 0: tile never crosses t
  const int sbase = row0 - t * NS;
  const int M     = lane & 15;
  const int hf    = (lane >> 4) * 8;
  const int s     = sbase + M;
  const float* __restrict__ xrow  = x + ((size_t)t * NS + s) * 6;
  const float* __restrict__ xcrow = xc + s * NG;
  _Float16* S = stage[wave];
  const int col = lane & 15;

  // -------- MLP1 (fcT1): input [LAI, xc], K=33 padded to 64 --------
  {
    auto inA = [&](int k) -> float {
      if (k == 0) return xrow[5];            // LAI
      if (k <= NG) return xcrow[k - 1];
      return 0.0f;
    };
    v16h a0, a1;
#pragma unroll
    for (int i = 0; i < 8; ++i) {
      a0[i]     = (_Float16)inA(hf + i);
      a0[8 + i] = (_Float16)inA(16 + hf + i);
      a1[i]     = (_Float16)inA(32 + hf + i);
      a1[8 + i] = (_Float16)inA(48 + hf + i);
    }
    v8f acc[2]; acc[0] = {}; acc[1] = {};
    for (int kk = 0; kk < 8; ++kk) {
#pragma unroll
      for (int sub = 0; sub < 2; ++sub) {
        const int n1 = kk * 2 + sub;
        v8f c = {};
        c = __builtin_amdgcn_wmma_f32_16x16x32_f16(false, a0, false,
              load_frag_packed(B1a, n1 * 2 + 0), (short)0, c, false, false);
        c = __builtin_amdgcn_wmma_f32_16x16x32_f16(false, a1, false,
              load_frag_packed(B1a, n1 * 2 + 1), (short)0, c, false, false);
        const float bias = b1a[n1 * 16 + col];
#pragma unroll
        for (int r = 0; r < 8; ++r) {
          const int Mr = r + hf;  // r + 8*half
          S[Mr * 32 + sub * 16 + col] = (_Float16)fast_tanh(c[r] + bias);
        }
      }
      asm volatile("s_wait_dscnt 0" ::: "memory");
      const v16h af = load_a_stage(S);
#pragma unroll
      for (int n2 = 0; n2 < 2; ++n2)
        acc[n2] = __builtin_amdgcn_wmma_f32_16x16x32_f16(false, af, false,
                    load_frag_packed(B2a, kk * 2 + n2), (short)0, acc[n2],
                    false, false);
    }
#pragma unroll
    for (int n2 = 0; n2 < 2; ++n2) {
      const int oc = col + n2 * 16;
      const float bias = b2a[oc];
#pragma unroll
      for (int r = 0; r < 8; ++r)
        v1ws[(size_t)(row0 + r + hf) * 32 + oc] = acc[n2][r] + bias;
    }
  }

  // -------- MLP2 (fcT2): input [R, T1, T2, xc], K=35 padded to 64 --------
  {
    auto inA = [&](int k) -> float {
      if (k == 0) return xrow[4];            // R
      if (k == 1) return xrow[2];            // T1
      if (k == 2) return xrow[3];            // T2
      if (k <= NG + 2) return xcrow[k - 3];
      return 0.0f;
    };
    v16h a0, a1;
#pragma unroll
    for (int i = 0; i < 8; ++i) {
      a0[i]     = (_Float16)inA(hf + i);
      a0[8 + i] = (_Float16)inA(16 + hf + i);
      a1[i]     = (_Float16)inA(32 + hf + i);
      a1[8 + i] = (_Float16)inA(48 + hf + i);
    }
    v8f acc = {};
    for (int kk = 0; kk < 8; ++kk) {
#pragma unroll
      for (int sub = 0; sub < 2; ++sub) {
        const int n1 = kk * 2 + sub;
        v8f c = {};
        c = __builtin_amdgcn_wmma_f32_16x16x32_f16(false, a0, false,
              load_frag_packed(B1b, n1 * 2 + 0), (short)0, c, false, false);
        c = __builtin_amdgcn_wmma_f32_16x16x32_f16(false, a1, false,
              load_frag_packed(B1b, n1 * 2 + 1), (short)0, c, false, false);
        const float bias = b1b[n1 * 16 + col];
#pragma unroll
        for (int r = 0; r < 8; ++r) {
          const int Mr = r + hf;
          S[Mr * 32 + sub * 16 + col] = (_Float16)fast_tanh(c[r] + bias);
        }
      }
      asm volatile("s_wait_dscnt 0" ::: "memory");
      const v16h af = load_a_stage(S);
      acc = __builtin_amdgcn_wmma_f32_16x16x32_f16(false, af, false,
              load_frag_packed(B2b, kk), (short)0, acc, false, false);
    }
    const float bias = b2b[col];
#pragma unroll
    for (int r = 0; r < 8; ++r)
      v2ws[(size_t)(row0 + r + hf) * 16 + col] = acc[r] + bias;
  }
}

// ---------------------------------------------------------------------------
// Kernel 4: fused temporal scan + causal routing conv + gate-weighted sum.
// one lane per (s,h); 16-lane shfl reduction over h; writes out[t,s].
// ---------------------------------------------------------------------------
__global__ __launch_bounds__(256) void scan_conv_kernel(
    const float* __restrict__ x, const float* __restrict__ v1ws,
    const float* __restrict__ v2ws, const float* __restrict__ wbuf,
    const float* __restrict__ wRbuf, float* __restrict__ out) {
  const int g = blockIdx.x * blockDim.x + threadIdx.x;
  const int s = g >> 4;
  const int h = g & 15;
  if (s >= NS) return;

  const float* wr = wbuf + s * (NH * 10);
  const float k1  = sigm(wr[1 * NH + h]);
  const float k2  = sigm(wr[2 * NH + h]);
  const float k23 = sigm(wr[3 * NH + h]);
  const float k3  = sigm(wr[4 * NH + h]) * 0.1f;
  const float gl  = __expf(wr[5 * NH + h]) * 2.0f;
  const float qb  = fmaxf(0.0f, wr[7 * NH + h]);
  const float ge1 = fmaxf(0.0f, wr[8 * NH + h]);
  const float ge2 = fmaxf(0.0f, wr[9 * NH + h]);
  // softmax over h for ga (stable, within 16-lane group)
  float ga;
  {
    const float z = wr[6 * NH + h];
    float m = z;
    m = fmaxf(m, __shfl_xor(m, 1, 16));
    m = fmaxf(m, __shfl_xor(m, 2, 16));
    m = fmaxf(m, __shfl_xor(m, 4, 16));
    m = fmaxf(m, __shfl_xor(m, 8, 16));
    const float e = __expf(z - m);
    float es = e;
    es += __shfl_xor(es, 1, 16);
    es += __shfl_xor(es, 2, 16);
    es += __shfl_xor(es, 4, 16);
    es += __shfl_xor(es, 8, 16);
    ga = e / es;
  }
  // routing weights, reversed to lag order: lag-d weight = relu(wR[s, h*NR + NR-1-d])
  float rw[NR];
#pragma unroll
  for (int d = 0; d < NR; ++d)
    rw[d] = fmaxf(0.0f, wRbuf[s * (NH * NR) + h * NR + (NR - 1 - d)]);

  float S0 = 0.f, Sv = 0.f, S2 = 0.f, S3 = 0.f;
  float q[NR];
#pragma unroll
  for (int d = 0; d < NR; ++d) q[d] = 0.f;

  for (int t = 0; t < NT; ++t) {
    const size_t row = (size_t)t * NS + s;
    const float* xr = x + row * 6;
    if (t + 1 < NT) {                         // cover latency of next forcing row
      __builtin_prefetch(xr + (size_t)NS * 6, 0, 1);
      __builtin_prefetch(v1ws + (row + NS) * 32 + h, 0, 1);
    }
    const float P = xr[0], E = xr[1], T1 = xr[2], T2 = xr[3];
    // snow/rain partition vp
    const float den = T2 - T1;
    float ratio = (T1 + T2) / (den == 0.0f ? 1.0f : den);
    ratio = fminf(1.0f, fmaxf(-1.0f, ratio));
    if (T1 >= 0.0f || T2 <= 0.0f) ratio = 0.0f;
    float vp = 1.0f - acosf(ratio) * (1.0f / 3.1415f);
    if (T1 >= 0.0f) vp = 1.0f;
    if (T2 <= 0.0f) vp = 0.0f;

    const float Ps = P * (1.0f - vp);
    const float Pl = P * vp;
    const float vi  = hsig(v1ws[row * 32 + h]);
    const float vkv = hsig(v1ws[row * 32 + NH + h]);
    const float vm  = __expf(2.0f * v2ws[row * 16 + h]);
    const float Pl1 = Pl * (1.0f - vi);
    const float Pl2 = Pl * vi;
    const float Ev1 = E * ge1;
    const float Ev2 = E * ge2;

    const float H0  = S0 + Ps;
    const float qSm = fminf(H0, vm);
    const float Hv  = fmaxf(0.0f, Sv + Pl1 - Ev1);
    const float qv  = Sv * vkv;
    const float H2  = fmaxf(0.0f, S2 + qSm + qv - Ev2 + Pl2);
    const float x1  = H2 - gl;
    const float Q1  = (x1 > 0.0f) ? __powf(x1, k1) : 0.0f;   // x1>0: fast path valid
    const float q2  = fminf(H2, gl) * k2;
    const float Q2  = q2 * (1.0f - k23);
    const float H3  = S3 + q2 * k23;
    const float Q3  = H3 * k3 + qb;
    S0 = H0 - qSm; Sv = Hv - qv; S2 = H2 - Q1 - q2; S3 = H3 - Q3;

    // fused causal conv (register ring) + gate
    const float qt = Q1 + Q2 + Q3;
#pragma unroll
    for (int d = NR - 1; d > 0; --d) q[d] = q[d - 1];
    q[0] = qt;
    float acc = 0.0f;
#pragma unroll
    for (int d = 0; d < NR; ++d) acc += rw[d] * q[d];
    acc *= ga;
    acc += __shfl_xor(acc, 1, 16);
    acc += __shfl_xor(acc, 2, 16);
    acc += __shfl_xor(acc, 4, 16);
    acc += __shfl_xor(acc, 8, 16);
    if (h == 0) out[row] = acc;
  }
}

// ---------------------------------------------------------------------------
extern "C" void kernel_launch(void* const* d_in, const int* in_sizes, int n_in,
                              void* d_out, int out_size, void* d_ws, size_t ws_size,
                              hipStream_t stream) {
  const float* x      = (const float*)d_in[0];
  const float* xc     = (const float*)d_in[1];
  const float* fcW1   = (const float*)d_in[2];
  const float* fcb1   = (const float*)d_in[3];
  const float* fcW2   = (const float*)d_in[4];
  const float* fcb2   = (const float*)d_in[5];
  const float* fcRW1  = (const float*)d_in[6];
  const float* fcRb1  = (const float*)d_in[7];
  const float* fcRW2  = (const float*)d_in[8];
  const float* fcRb2  = (const float*)d_in[9];
  const float* fcT1W1 = (const float*)d_in[10];
  const float* fcT1b1 = (const float*)d_in[11];
  const float* fcT1W2 = (const float*)d_in[12];
  const float* fcT1b2 = (const float*)d_in[13];
  const float* fcT2W1 = (const float*)d_in[14];
  const float* fcT2b1 = (const float*)d_in[15];
  const float* fcT2W2 = (const float*)d_in[16];
  const float* fcT2b2 = (const float*)d_in[17];

  char* ws = (char*)d_ws;
  size_t off = 0;
  float* v1ws = (float*)(ws + off); off += (size_t)NT * NS * 32 * 4;
  float* v2ws = (float*)(ws + off); off += (size_t)NT * NS * 16 * 4;
  float* wbuf  = (float*)(ws + off); off += (size_t)NS * NH * 10 * 4;
  float* wRbuf = (float*)(ws + off); off += (size_t)NS * NH * NR * 4;
  _Float16* B1a = (_Float16*)(ws + off); off += 32 * 512 * 2;
  _Float16* B2a = (_Float16*)(ws + off); off += 16 * 512 * 2;
  _Float16* B1b = (_Float16*)(ws + off); off += 32 * 512 * 2;
  _Float16* B2b = (_Float16*)(ws + off); off += 8 * 512 * 2;

  hipLaunchKernelGGL(prep_weights_kernel, dim3(64), dim3(256), 0, stream,
                     fcT1W1, fcT1W2, fcT2W1, fcT2W2, B1a, B2a, B1b, B2b);
  hipLaunchKernelGGL(params_mlp_kernel, dim3(NS), dim3(256), 0, stream,
                     xc, fcW1, fcb1, fcW2, fcb2, fcRW1, fcRb1, fcRW2, fcRb2,
                     wbuf, wRbuf);
  const int ntiles = (NT * NS) / 16;                 // 91250 exact
  hipLaunchKernelGGL(fused_mlp_kernel, dim3((ntiles + 7) / 8), dim3(256), 0, stream,
                     x, xc, B1a, fcT1b1, B2a, fcT1b2, B1b, fcT2b1, B2b, fcT2b2,
                     v1ws, v2ws);
  hipLaunchKernelGGL(scan_conv_kernel, dim3((NS * NH) / 256), dim3(256), 0, stream,
                     x, v1ws, v2ws, wbuf, wRbuf, (float*)d_out);
}
